// LinearAttention_28793460753011
// MI455X (gfx1250) — matrix operand order repaired and checked by reading, not compiled, not run on hardware
//
#include <hip/hip_runtime.h>

// Linformer attention for MI455X (gfx1250), bf16 WMMA path.
// B=8, L=S=4096, H=16, E=64, PK=256.

#define B_  8
#define L_  4096
#define H_  16
#define E_  64
#define PK_ 256

typedef __attribute__((ext_vector_type(16))) __bf16 v16bf;
typedef __attribute__((ext_vector_type(8)))  float  v8f;
typedef __attribute__((__vector_size__(16))) int    v4i;

// Fragment viewed either as the WMMA operand vector or as raw 16B chunks.
union frag16 {
  v16bf v;
  uint4 q[2];
  unsigned u[8];
};

// Force a single v_cvt_pk_bf16_f32.
__device__ __forceinline__ unsigned pk_bf16(float lo, float hi) {
  unsigned o;
  asm("v_cvt_pk_bf16_f32 %0, %1, %2" : "=v"(o) : "v"(lo), "v"(hi));
  return o;
}

__device__ __forceinline__ v8f wmma_bf16(v16bf a, v16bf b, v8f c) {
  return __builtin_amdgcn_wmma_f32_16x16x32_bf16(
      false, a, false, b, (short)0, c, false, false);
}

// ---------------------------------------------------------------------------
// Kernel 1: P = W[256x4096] @ X_bh[4096x64], bf16 in LDS, double-buffered.
// grid = (B*H, 2 M-halves), block = 256 (8 waves).
// mode 0: row-major bf16 out [bh][m][e]        (k_proj, consumed transposed)
// mode 1: pair-packed uint out [bh][e][m/2]    (v_proj, consumed as B matrix)
// ---------------------------------------------------------------------------
__global__ __launch_bounds__(256)
void linf_proj_kernel(const float* __restrict__ X,   // [B, S, H, E]
                      const float* __restrict__ W,   // [PK, S]
                      void* __restrict__ Pout, const int mode)
{
  __shared__ __bf16   ldsW[2][128 * 32];  // 2 x 8 KB, row-major bf16
  __shared__ unsigned ldsX[2][64 * 16];   // 2 x 4 KB, [e-col][k-pair] packed

  const int bh   = blockIdx.x;
  const int b    = bh >> 4, h = bh & 15;
  const int M0   = blockIdx.y * 128;
  const int tid  = threadIdx.x;
  const int wave = tid >> 5, lane = tid & 31;
  const int half = lane >> 4, r = lane & 15;

  const float* Xbh = X + (size_t)b * (L_ * H_ * E_) + h * E_;

  // X staging task: this thread packs k rows (2*xk2, 2*xk2+1), cols xnc..xnc+3
  const int xk2 = tid & 15;
  const int xnc = (tid >> 4) << 2;

  v8f acc[4];
#pragma unroll
  for (int nt = 0; nt < 4; ++nt)
#pragma unroll
    for (int i = 0; i < 8; ++i) acc[nt][i] = 0.f;

  float4 wreg[4];
  float4 xr0, xr1;

  auto stage_load = [&](int k0) {
#pragma unroll
    for (int i = 0; i < 4; ++i) {
      int f = tid + i * 256;
      int row = f >> 3, c4 = (f & 7) << 2;
      wreg[i] = *(const float4*)(W + (size_t)(M0 + row) * L_ + k0 + c4);
    }
    xr0 = *(const float4*)(Xbh + (size_t)(k0 + 2 * xk2)     * (H_ * E_) + xnc);
    xr1 = *(const float4*)(Xbh + (size_t)(k0 + 2 * xk2 + 1) * (H_ * E_) + xnc);
  };
  auto stage_store = [&](int bs) {
#pragma unroll
    for (int i = 0; i < 4; ++i) {
      int f = tid + i * 256;
      int row = f >> 3, c4 = (f & 7) << 2;
      uint2 u;
      u.x = pk_bf16(wreg[i].x, wreg[i].y);
      u.y = pk_bf16(wreg[i].z, wreg[i].w);
      *(uint2*)&ldsW[bs][row * 32 + c4] = u;
    }
    const float* a0 = (const float*)&xr0;
    const float* a1 = (const float*)&xr1;
#pragma unroll
    for (int j = 0; j < 4; ++j)
      ldsX[bs][(xnc + j) * 16 + xk2] = pk_bf16(a0[j], a1[j]);
  };

  stage_load(0);
  stage_store(0);
  __syncthreads();

  for (int it = 0; it < L_ / 32; ++it) {
    if (it + 1 < L_ / 32) stage_load((it + 1) * 32);
    const int bs = it & 1;

    // Load ALL fragments first, then issue WMMAs back-to-back.
    frag16 a, bb[4];
    const char* wrow = (const char*)&ldsW[bs][(wave * 16 + r) * 32];
    a.q[0] = *(const uint4*)(wrow + half * 16);
    a.q[1] = *(const uint4*)(wrow + 32 + half * 16);
#pragma unroll
    for (int nt = 0; nt < 4; ++nt) {
      const unsigned* xcol = &ldsX[bs][(nt * 16 + r) * 16];
      bb[nt].q[0] = *(const uint4*)(xcol + half * 4);
      bb[nt].q[1] = *(const uint4*)(xcol + 8 + half * 4);
    }
#pragma unroll
    for (int nt = 0; nt < 4; ++nt)
      acc[nt] = wmma_bf16(a.v, bb[nt].v, acc[nt]);

    if (it + 1 < L_ / 32) stage_store((it + 1) & 1);
    __syncthreads();
  }

  // C layout: VGPR rr, lanes 0-15 -> M=rr, lanes 16-31 -> M=rr+8; N=lane%16
  const int mb = M0 + wave * 16 + half * 8;  // this lane's 8 consecutive rows
  if (mode == 0) {
    __bf16* out = (__bf16*)Pout + (size_t)bh * (PK_ * E_);
#pragma unroll
    for (int nt = 0; nt < 4; ++nt)
#pragma unroll
      for (int rr = 0; rr < 8; ++rr)
        out[(mb + rr) * E_ + nt * 16 + r] = (__bf16)acc[nt][rr];
  } else {
    // [e][m2] pair-packed: uint at e*(PK/2) + m/2 holds rows (m, m+1)
    unsigned* out = (unsigned*)Pout + (size_t)bh * ((PK_ / 2) * E_);
#pragma unroll
    for (int nt = 0; nt < 4; ++nt)
#pragma unroll
      for (int rr2 = 0; rr2 < 4; ++rr2)
        out[(nt * 16 + r) * (PK_ / 2) + (mb >> 1) + rr2] =
            pk_bf16(acc[nt][2 * rr2], acc[nt][2 * rr2 + 1]);
  }
}

// ---------------------------------------------------------------------------
// Kernel 2: scores = Q @ kproj^T ; softmax ; out = A @ vproj
// grid = (L/128, B*H), block = 256 (8 waves, one 16-row stripe per wave).
// ---------------------------------------------------------------------------
__global__ __launch_bounds__(256)
void linf_attn_kernel(const float* __restrict__ Q,      // [B, L, H, E]
                      const __bf16* __restrict__ KP,    // [bh][m][e] bf16
                      const unsigned* __restrict__ VPT, // [bh][e][m2] packed
                      float* __restrict__ Out)          // [B, L, H, E]
{
  __shared__ __bf16   kp[PK_ * E_];   // 32 KB; later reused as P-stage
  __shared__ unsigned vpt[E_ * 128];  // 32 KB

  const int l0   = blockIdx.x * 128;
  const int bh   = blockIdx.y;
  const int b    = bh >> 4, h = bh & 15;
  const int tid  = threadIdx.x;
  const int wave = tid >> 5, lane = tid & 31;
  const int half = lane >> 4, r = lane & 15;

  // Async DMA of both projections into LDS (no VGPR round-trip; ASYNCcnt).
  {
    __attribute__((address_space(1))) v4i* gk =
        (__attribute__((address_space(1))) v4i*)(KP + (size_t)bh * (PK_ * E_));
    __attribute__((address_space(1))) v4i* gv =
        (__attribute__((address_space(1))) v4i*)(VPT + (size_t)bh * (E_ * 128));
    __attribute__((address_space(3))) v4i* lk =
        (__attribute__((address_space(3))) v4i*)kp;
    __attribute__((address_space(3))) v4i* lv =
        (__attribute__((address_space(3))) v4i*)vpt;
#pragma unroll
    for (int i = 0; i < 8; ++i) {
      int idx = tid + i * 256;  // 16B elements
      __builtin_amdgcn_global_load_async_to_lds_b128(gk + idx, lk + idx, 0, 0);
      __builtin_amdgcn_global_load_async_to_lds_b128(gv + idx, lv + idx, 0, 0);
    }
  }

  // Overlap the async DMA with Q A-fragment loads (global -> VGPR, pack bf16)
  const float* Qb = Q + (size_t)b * (L_ * H_ * E_) + h * E_;
  const float* Qrow = Qb + (size_t)(l0 + wave * 16 + r) * (H_ * E_);
  frag16 qa[2];
#pragma unroll
  for (int kt = 0; kt < 2; ++kt) {
    float4 f0 = *(const float4*)(Qrow + kt * 32 + half * 8);
    float4 f1 = *(const float4*)(Qrow + kt * 32 + half * 8 + 4);
    float4 f2 = *(const float4*)(Qrow + kt * 32 + 16 + half * 8);
    float4 f3 = *(const float4*)(Qrow + kt * 32 + 16 + half * 8 + 4);
    qa[kt].u[0] = pk_bf16(f0.x, f0.y); qa[kt].u[1] = pk_bf16(f0.z, f0.w);
    qa[kt].u[2] = pk_bf16(f1.x, f1.y); qa[kt].u[3] = pk_bf16(f1.z, f1.w);
    qa[kt].u[4] = pk_bf16(f2.x, f2.y); qa[kt].u[5] = pk_bf16(f2.z, f2.w);
    qa[kt].u[6] = pk_bf16(f3.x, f3.y); qa[kt].u[7] = pk_bf16(f3.z, f3.w);
  }

  // Drain this wave's async copies, then block-wide barrier.
  asm volatile("s_wait_asynccnt 0x0" ::: "memory");
  __syncthreads();

  // scores stripe: [16 rows x 256 cols] in 16 accumulator tiles
  v8f s[16];
#pragma unroll
  for (int nt = 0; nt < 16; ++nt)
#pragma unroll
    for (int i = 0; i < 8; ++i) s[nt][i] = 0.f;

#pragma unroll
  for (int g = 0; g < 4; ++g) {
#pragma unroll
    for (int kt = 0; kt < 2; ++kt) {
      frag16 bb[4];
#pragma unroll
      for (int j = 0; j < 4; ++j) {
        const char* krow = (const char*)&kp[((g * 4 + j) * 16 + r) * E_];
        bb[j].q[0] = *(const uint4*)(krow + kt * 64 + half * 16);
        bb[j].q[1] = *(const uint4*)(krow + kt * 64 + half * 16 + 32);
      }
#pragma unroll
      for (int j = 0; j < 4; ++j)
        s[g * 4 + j] = wmma_bf16(qa[kt].v, bb[j].v, s[g * 4 + j]);
    }
  }

  // In-register softmax; row M = rr + 8*(lane/16) spans one 16-lane half group
  const float sc2 = 0.125f * 1.4426950408889634f;  // (1/sqrt(E)) * log2(e)
  float rsum[8];
#pragma unroll
  for (int rr = 0; rr < 8; ++rr) {
    float m = s[0][rr];
#pragma unroll
    for (int nt = 1; nt < 16; ++nt) m = fmaxf(m, s[nt][rr]);
#pragma unroll
    for (int d = 1; d < 16; d <<= 1) m = fmaxf(m, __shfl_xor(m, d, 32));
    const float mm = m * sc2;
    float sum = 0.f;
#pragma unroll
    for (int nt = 0; nt < 16; ++nt) {
      float e = exp2f(__builtin_fmaf(s[nt][rr], sc2, -mm));
      s[nt][rr] = e;
      sum += e;
    }
#pragma unroll
    for (int d = 1; d < 16; d <<= 1) sum += __shfl_xor(sum, d, 32);
    rsum[rr] = sum;
  }

  __syncthreads();  // all waves done reading kp -> reuse as P staging

  __bf16* ps = kp + wave * (16 * 128);  // 4 KB wave-private stripe
  v8f o[4];
#pragma unroll
  for (int nt = 0; nt < 4; ++nt)
#pragma unroll
    for (int i = 0; i < 8; ++i) o[nt][i] = 0.f;

#pragma unroll
  for (int c = 0; c < 2; ++c) {
    // Stage P cols c*128..+127 row-major bf16 (C-layout -> memory)
#pragma unroll
    for (int ntl = 0; ntl < 8; ++ntl) {
      int nt = c * 8 + ntl;
#pragma unroll
      for (int rr = 0; rr < 8; ++rr)
        ps[(rr + half * 8) * 128 + ntl * 16 + r] = (__bf16)s[nt][rr];
    }
    // LDS ops are in-order within a wave: loads below see the stores above.
#pragma unroll
    for (int ktl = 0; ktl < 4; ++ktl) {
      int kt = c * 4 + ktl;
      frag16 a, bb[4];
      const char* prow = (const char*)&ps[r * 128];
      a.q[0] = *(const uint4*)(prow + ktl * 64 + half * 16);
      a.q[1] = *(const uint4*)(prow + ktl * 64 + half * 16 + 32);
#pragma unroll
      for (int nt2 = 0; nt2 < 4; ++nt2) {
        const unsigned* vcol = &vpt[(nt2 * 16 + r) * 128];
        bb[nt2].q[0] = *(const uint4*)(vcol + kt * 16 + half * 4);
        bb[nt2].q[1] = *(const uint4*)(vcol + kt * 16 + half * 4 + 8);
      }
#pragma unroll
      for (int nt2 = 0; nt2 < 4; ++nt2)
        o[nt2] = wmma_bf16(a.v, bb[nt2].v, o[nt2]);
    }
  }

  // Deferred 1/rowsum normalization + f32 store
  float* Ob = Out + (size_t)b * (L_ * H_ * E_) + h * E_;
#pragma unroll
  for (int rr = 0; rr < 8; ++rr) {
    float inv = 1.f / rsum[rr];
    int grow = l0 + wave * 16 + rr + half * 8;
#pragma unroll
    for (int nt = 0; nt < 4; ++nt)
      Ob[(size_t)grow * (H_ * E_) + nt * 16 + r] = o[nt][rr] * inv;
  }
}

// ---------------------------------------------------------------------------
extern "C" void kernel_launch(void* const* d_in, const int* in_sizes, int n_in,
                              void* d_out, int out_size, void* d_ws, size_t ws_size,
                              hipStream_t stream) {
  (void)in_sizes; (void)n_in; (void)out_size; (void)ws_size;
  const float* q  = (const float*)d_in[0];
  const float* k  = (const float*)d_in[1];
  const float* v  = (const float*)d_in[2];
  const float* Wk = (const float*)d_in[3];
  const float* Wv = (const float*)d_in[4];

  const size_t proj_bytes = (size_t)(B_ * H_) * PK_ * E_ * 2;  // 4 MB each
  void* kproj = d_ws;
  void* vproj = (char*)d_ws + proj_bytes;

  dim3 gp(B_ * H_, 2), bp(256);
  linf_proj_kernel<<<gp, bp, 0, stream>>>(k, Wk, kproj, 0);
  linf_proj_kernel<<<gp, bp, 0, stream>>>(v, Wv, vproj, 1);

  dim3 ga(L_ / 128, B_ * H_), ba(256);
  linf_attn_kernel<<<ga, ba, 0, stream>>>(q, (const __bf16*)kproj,
                                          (const unsigned*)vproj,
                                          (float*)d_out);
}
